// MultiHeadAttention_10746008174639
// MI455X (gfx1250) — matrix operand order repaired and checked by reading, not compiled, not run on hardware
//
#include <hip/hip_runtime.h>
#include <hip/hip_bf16.h>

// Problem constants (match reference)
#define BB    4
#define SS    2048
#define EE    1024
#define HH    16
#define HD    64
#define MTOT  (BB*SS)     // 8192
#define NQKV  (3*EE)      // 3072

typedef __bf16 bf16_t;
typedef __attribute__((ext_vector_type(16))) __bf16 v16bf;
typedef __attribute__((ext_vector_type(8)))  float  v8f;

__device__ __forceinline__ bf16_t f2bf(float f) { return (bf16_t)f; }

// ---- CDNA5 async copy: global -> LDS (ASYNCcnt-tracked) --------------------

// Generic pointers to __shared__ data are {aperture_hi, lds_offset}; the low
// 32 bits are the LDS byte offset the async instruction's VDST VGPR wants.
__device__ __forceinline__ unsigned lds_off(const void* p) {
  return (unsigned)(unsigned long long)p;
}

__device__ __forceinline__ void async_copy_b128(unsigned lds_byte_off,
                                                const void* gaddr) {
  asm volatile("global_load_async_to_lds_b128 %0, %1, off"
               :: "v"(lds_byte_off), "v"(gaddr)
               : "memory");
}

__device__ __forceinline__ void wait_async_le4() {
  asm volatile("s_wait_asynccnt 4" ::: "memory");
}
__device__ __forceinline__ void wait_async_le0() {
  asm volatile("s_wait_asynccnt 0" ::: "memory");
}

// ---- WMMA fragment helpers (contiguous k order, same for A and B) ------------------

// A-matrix 16x32 bf16: lane&15 = M row; lanes0-15 hold K 0..15,
// lanes16-31 hold K 16..31.  base -> element (m=0,k=0).
__device__ __forceinline__ v16bf load_a16x32(const bf16_t* base, int stride) {
  const int lane = threadIdx.x & 31;
  const int m  = lane & 15;
  const int kh = (lane >> 4) << 4;       // 0 or 16
  const bf16_t* p = base + m * stride + kh;
  v16bf f;
#pragma unroll
  for (int i = 0; i < 16; ++i) f[i] = p[i];
  return f;
}

// B-matrix 32x16 bf16 from a [N][K]-stored (transposed) array:
// lane&15 = N col; lanes0-15 hold K 0..15, lanes16-31 hold K 16..31.
// baseT -> element (n=0,k=0).
__device__ __forceinline__ v16bf load_b32x16(const bf16_t* baseT, int stride) {
  const int lane = threadIdx.x & 31;
  const int n  = lane & 15;
  const int kb = (lane >> 4) << 4;       // 0 or 16
  const bf16_t* p = baseT + n * stride + kb;
  v16bf f;
#pragma unroll
  for (int i = 0; i < 16; ++i) f[i] = p[i];
  return f;
}

__device__ __forceinline__ v8f wmma_bf16(v16bf a, v16bf b, v8f c) {
  return __builtin_amdgcn_wmma_f32_16x16x32_bf16(
      /*neg_a=*/false, a, /*neg_b=*/false, b,
      /*c_mod=*/(short)0, c, /*reuse_a=*/false, /*reuse_b=*/false);
}

// ---- Pack kernels ----------------------------------------------------------

__global__ void cvt_f32_bf16_kernel(const float* __restrict__ src,
                                    bf16_t* __restrict__ dst, int n) {
  for (int i = blockIdx.x * blockDim.x + threadIdx.x; i < n;
       i += gridDim.x * blockDim.x)
    dst[i] = f2bf(src[i]);
}

// w: [K][N] f32 row-major  ->  wT: [N][K] bf16
__global__ void transpose_w_kernel(const float* __restrict__ w,
                                   bf16_t* __restrict__ wT, int K, int N) {
  int total = K * N;
  for (int i = blockIdx.x * blockDim.x + threadIdx.x; i < total;
       i += gridDim.x * blockDim.x) {
    int k = i / N, n = i - k * N;
    wT[(size_t)n * K + k] = f2bf(w[i]);
  }
}

// ---- QKV GEMM:  qkv = x @ W_attn + b_attn, scattered to Q/K/V^T ------------
// xb: [8192][1024] bf16, wT: [3072][1024] bf16.
// 256 thr = 8 waves; wave tile 32x64; block tile 128x128.
__global__ void __launch_bounds__(256)
qkv_gemm_kernel(const bf16_t* __restrict__ xb, const bf16_t* __restrict__ wT,
                const float* __restrict__ bias,
                bf16_t* __restrict__ qbuf,   // [B*H][S][64]
                bf16_t* __restrict__ kbuf,   // [B*H][S][64]
                bf16_t* __restrict__ vTbuf)  // [B*H][64][S]
{
  const int wid  = threadIdx.x >> 5;
  const int row0 = blockIdx.x * 128 + (wid >> 1) * 32;
  const int col0 = blockIdx.y * 128 + (wid & 1) * 64;

  v8f acc[2][4] = {};
  for (int k = 0; k < EE; k += 32) {
    v16bf a0 = load_a16x32(xb + (size_t)(row0 +  0) * EE + k, EE);
    v16bf a1 = load_a16x32(xb + (size_t)(row0 + 16) * EE + k, EE);
#pragma unroll
    for (int j = 0; j < 4; ++j) {
      v16bf b = load_b32x16(wT + (size_t)(col0 + j * 16) * EE + k, EE);
      acc[0][j] = wmma_bf16(a0, b, acc[0][j]);
      acc[1][j] = wmma_bf16(a1, b, acc[1][j]);
    }
  }

  const int lane = threadIdx.x & 31;
  const int n  = lane & 15;
  const int hi = lane >> 4;
#pragma unroll
  for (int i = 0; i < 2; ++i) {
#pragma unroll
    for (int j = 0; j < 4; ++j) {
      const int gcol = col0 + j * 16 + n;
      const float bv = bias[gcol];
      const int sel = gcol >> 10;        // 0=Q 1=K 2=V
      const int e   = gcol & 1023;
      const int h   = e >> 6;
      const int d   = e & 63;
#pragma unroll
      for (int r = 0; r < 8; ++r) {
        const int grow = row0 + i * 16 + r + hi * 8;
        const int bidx = grow >> 11;     // /2048
        const int s    = grow & 2047;
        const int bh   = bidx * HH + h;
        const bf16_t v = f2bf(acc[i][j][r] + bv);
        if (sel == 0)      qbuf[((size_t)bh * SS + s) * HD + d] = v;
        else if (sel == 1) kbuf[((size_t)bh * SS + s) * HD + d] = v;
        else               vTbuf[((size_t)bh * HD + d) * SS + s] = v;
      }
    }
  }
}

// ---- Flash attention with async-LDS double-buffered K/V staging ------------
// block = 128 thr (4 waves), grid = (S/64, B*H). Wave owns 16 query rows.
// Per 32-key tile: K tile [32][64] bf16 (4KB, contiguous in global) and
// V^T tile [64][32] bf16 (4KB) staged once per block via
// global_load_async_to_lds_b128 (4 ops/wave/tile), double-buffered.
__global__ void __launch_bounds__(128)
attn_kernel(const bf16_t* __restrict__ qbuf,   // [BH][S][64]
            const bf16_t* __restrict__ kbuf,   // [BH][S][64]
            const bf16_t* __restrict__ vTbuf,  // [BH][64][S]
            bf16_t* __restrict__ obuf)         // [B][S][E]
{
  __shared__ __align__(16) bf16_t kT[2][32][64];   // [buf][key][d]
  __shared__ __align__(16) bf16_t vT[2][64][32];   // [buf][d][key]
  __shared__ __align__(16) bf16_t pbuf[4][16][32]; // per-wave P tile

  const int wid  = threadIdx.x >> 5;
  const int lane = threadIdx.x & 31;
  const int n    = lane & 15;
  const int hi   = lane >> 4;

  const int bh    = blockIdx.y;
  const int bidx  = bh >> 4;
  const int h     = bh & 15;
  const int q0    = blockIdx.x * 64;
  const int qrow0 = q0 + wid * 16;

  const bf16_t* qbase = qbuf  + ((size_t)bh * SS + qrow0) * HD;
  const bf16_t* kbase = kbuf  + (size_t)bh * SS * HD;
  const bf16_t* vbase = vTbuf + (size_t)bh * HD * SS;

  const v16bf qf0 = load_a16x32(qbase +  0, HD);   // d 0..31
  const v16bf qf1 = load_a16x32(qbase + 32, HD);   // d 32..63

  v8f o[4] = {};
  float mrow[8], lrow[8];
#pragma unroll
  for (int r = 0; r < 8; ++r) { mrow[r] = -1e30f; lrow[r] = 0.f; }

  const float scale = 0.125f;                      // 1/sqrt(64)
  const int ntiles = (q0 + 64) / 32;               // uniform over the block

  // Per-tile async stage: this wave copies bytes [wid*1024, wid*1024+1024)
  // of each 4KB tile; lane handles a 16B chunk, two 512B instructions each.
  const int so = wid * 1024 + lane * 16;           // 16B-aligned flat offset
  auto stage_tile = [&](int tile, int buf) {
    const char* gk = (const char*)(kbase + (size_t)tile * 32 * HD); // contiguous 4KB
    const unsigned lk = lds_off(&kT[buf][0][0]);
    async_copy_b128(lk + so,       gk + so);
    async_copy_b128(lk + so + 512, gk + so + 512);
    // V^T tile rows: 64B per d-row, global row stride SS*2 bytes
    const unsigned lv = lds_off(&vT[buf][0][0]);
    const char* gv = (const char*)vbase + (size_t)tile * 32 * 2;    // + k0 cols
    const int r0 = so >> 6, c0 = so & 63;
    async_copy_b128(lv + so, gv + (size_t)r0 * (SS * 2) + c0);
    const int o2 = so + 512;
    const int r2 = o2 >> 6, c2 = o2 & 63;
    async_copy_b128(lv + o2, gv + (size_t)r2 * (SS * 2) + c2);
  };

  stage_tile(0, 0);

  for (int i = 0; i < ntiles; ++i) {
    const int k0  = i * 32;
    const int cur = i & 1;
    const bool more = (i + 1) < ntiles;
    if (more) stage_tile(i + 1, cur ^ 1);
    if (more) wait_async_le4(); else wait_async_le0();  // tile i landed (in-order)
    __syncthreads();

    // S = Q K^T from LDS K tile (two 16x16 key tiles, chained over d=64)
    v8f s0 = {}, s1 = {};
    s0 = wmma_bf16(qf0, load_b32x16(&kT[cur][ 0][ 0], HD), s0);
    s0 = wmma_bf16(qf1, load_b32x16(&kT[cur][ 0][32], HD), s0);
    s1 = wmma_bf16(qf0, load_b32x16(&kT[cur][16][ 0], HD), s1);
    s1 = wmma_bf16(qf1, load_b32x16(&kT[cur][16][32], HD), s1);

    // causal mask + scale + row max (rows live in 16-lane halves)
    float tmax[8];
#pragma unroll
    for (int r = 0; r < 8; ++r) {
      const int qr = qrow0 + r + hi * 8;
      float v0 = (k0 + n      <= qr) ? s0[r] * scale : -1e30f;
      float v1 = (k0 + 16 + n <= qr) ? s1[r] * scale : -1e30f;
      s0[r] = v0; s1[r] = v1;
      tmax[r] = fmaxf(v0, v1);
    }
#pragma unroll
    for (int r = 0; r < 8; ++r)
#pragma unroll
      for (int off = 8; off >= 1; off >>= 1)
        tmax[r] = fmaxf(tmax[r], __shfl_xor(tmax[r], off, 16));

    // online softmax update
    float alpha[8];
#pragma unroll
    for (int r = 0; r < 8; ++r) {
      const float mnew = fmaxf(mrow[r], tmax[r]);
      alpha[r] = __expf(mrow[r] - mnew);
      mrow[r] = mnew;
      const float p0 = __expf(s0[r] - mnew);
      const float p1 = __expf(s1[r] - mnew);
      s0[r] = p0; s1[r] = p1;
      float rs = p0 + p1;
#pragma unroll
      for (int off = 8; off >= 1; off >>= 1) rs += __shfl_xor(rs, off, 16);
      lrow[r] = lrow[r] * alpha[r] + rs;
    }

    // P: D-layout -> LDS -> A-layout
#pragma unroll
    for (int r = 0; r < 8; ++r) {
      const int row = r + hi * 8;
      pbuf[wid][row][n]      = f2bf(s0[r]);
      pbuf[wid][row][16 + n] = f2bf(s1[r]);
    }
    __syncthreads();

#pragma unroll
    for (int t = 0; t < 4; ++t)
#pragma unroll
      for (int r = 0; r < 8; ++r) o[t][r] *= alpha[r];

    const v16bf pf = load_a16x32(&pbuf[wid][0][0], 32);
#pragma unroll
    for (int t = 0; t < 4; ++t) {
      v16bf vf = load_b32x16(&vT[cur][t * 16][0], 32);
      o[t] = wmma_bf16(pf, vf, o[t]);
    }
    __syncthreads();   // all reads of kT/vT[cur] done before it is re-staged
  }

  // epilogue: O / l  -> bf16 [B][S][E]
  bf16_t* orow = obuf + ((size_t)(bidx * SS + qrow0)) * EE + h * HD;
#pragma unroll
  for (int t = 0; t < 4; ++t)
#pragma unroll
    for (int r = 0; r < 8; ++r) {
      const int row = r + hi * 8;
      orow[(size_t)row * EE + t * 16 + n] = f2bf(o[t][r] / lrow[r]);
    }
}

// ---- Output projection: out = attn @ W_proj + b_proj (f32 out) -------------
__global__ void __launch_bounds__(256)
proj_gemm_kernel(const bf16_t* __restrict__ ab, const bf16_t* __restrict__ wT,
                 const float* __restrict__ bias, float* __restrict__ out) {
  const int wid  = threadIdx.x >> 5;
  const int row0 = blockIdx.x * 128 + (wid >> 1) * 32;
  const int col0 = blockIdx.y * 128 + (wid & 1) * 64;

  v8f acc[2][4] = {};
  for (int k = 0; k < EE; k += 32) {
    v16bf a0 = load_a16x32(ab + (size_t)(row0 +  0) * EE + k, EE);
    v16bf a1 = load_a16x32(ab + (size_t)(row0 + 16) * EE + k, EE);
#pragma unroll
    for (int j = 0; j < 4; ++j) {
      v16bf b = load_b32x16(wT + (size_t)(col0 + j * 16) * EE + k, EE);
      acc[0][j] = wmma_bf16(a0, b, acc[0][j]);
      acc[1][j] = wmma_bf16(a1, b, acc[1][j]);
    }
  }

  const int lane = threadIdx.x & 31;
  const int n  = lane & 15;
  const int hi = lane >> 4;
#pragma unroll
  for (int i = 0; i < 2; ++i)
#pragma unroll
    for (int j = 0; j < 4; ++j) {
      const int gcol = col0 + j * 16 + n;
      const float bv = bias[gcol];
#pragma unroll
      for (int r = 0; r < 8; ++r) {
        const int grow = row0 + i * 16 + r + hi * 8;
        out[(size_t)grow * EE + gcol] = acc[i][j][r] + bv;
      }
    }
}

// ---- Host launch -----------------------------------------------------------

extern "C" void kernel_launch(void* const* d_in, const int* in_sizes, int n_in,
                              void* d_out, int out_size, void* d_ws, size_t ws_size,
                              hipStream_t stream) {
  const float* x      = (const float*)d_in[0];
  const float* W_attn = (const float*)d_in[1];
  const float* b_attn = (const float*)d_in[2];
  const float* W_proj = (const float*)d_in[3];
  const float* b_proj = (const float*)d_in[4];
  float* out = (float*)d_out;

  char* ws = (char*)d_ws;
  const size_t szXB   = (size_t)MTOT * EE * sizeof(bf16_t);         // 16 MB
  const size_t szWA   = (size_t)NQKV * EE * sizeof(bf16_t);         //  6 MB
  const size_t szWP   = (size_t)EE * EE * sizeof(bf16_t);           //  2 MB
  const size_t szQKV  = (size_t)BB * HH * SS * HD * sizeof(bf16_t); // 16 MB each
  bf16_t* xb     = (bf16_t*)ws;                 ws += szXB;
  bf16_t* wattnT = (bf16_t*)ws;                 ws += szWA;
  bf16_t* wprojT = (bf16_t*)ws;                 ws += szWP;
  bf16_t* qbuf   = (bf16_t*)ws;                 ws += szQKV;
  bf16_t* kbuf   = (bf16_t*)ws;                 ws += szQKV;
  bf16_t* vTbuf  = (bf16_t*)ws;                 ws += szQKV;
  bf16_t* attnb  = (bf16_t*)ws;                 ws += szXB;

  cvt_f32_bf16_kernel<<<8192, 256, 0, stream>>>(x, xb, MTOT * EE);
  transpose_w_kernel<<<8192, 256, 0, stream>>>(W_attn, wattnT, EE, NQKV);
  transpose_w_kernel<<<4096, 256, 0, stream>>>(W_proj, wprojT, EE, EE);

  dim3 g1(MTOT / 128, NQKV / 128);
  qkv_gemm_kernel<<<g1, 256, 0, stream>>>(xb, wattnT, b_attn, qbuf, kbuf, vTbuf);

  dim3 g2(SS / 64, BB * HH);
  attn_kernel<<<g2, 128, 0, stream>>>(qbuf, kbuf, vTbuf, attnb);

  dim3 g3(MTOT / 128, EE / 128);
  proj_gemm_kernel<<<g3, 256, 0, stream>>>(attnb, wprojT, b_proj, out);
}
